// ConvGRU_8486855376939
// MI455X (gfx1250) — matrix operand order repaired and checked by reading, not compiled, not run on hardware
//
#include <hip/hip_runtime.h>
#include <hip/hip_bf16.h>

// ConvGRU on gfx1250: bf16 WMMA GEMMs (16x16x32) with fp32 accum,
// recurrent state h kept in fp32 VGPRs across all T=16 steps.
//
// Shapes: video [B=4,T=16,C=192,H=56,W=56] fp32, w_* [192,384], b_* [192].
// Per step per pixel-tile: Z/R GEMM (M=192x2 gates, K=384), then H GEMM.

#define CC     192
#define C2     384
#define HW_    3136
#define T_     16
#define B_     4
#define KT_    12              // K tiles of 32 over 384
#define MT_    12              // M tiles of 16 over 192
#define TILEN  32              // pixels per workgroup (2 WMMA N-columns)
#define BLOCK  192             // 6 wave32
#define NPIX   12544           // 4*56*56
#define GATE_STRIDE (MT_*KT_*32*16)   // 73728 bf16 elems per packed gate

typedef __attribute__((ext_vector_type(16))) __bf16 v16bf;
typedef __attribute__((ext_vector_type(8)))  float  v8f;

#define AS1 __attribute__((address_space(1)))
typedef unsigned long long uptr;

// Global-AS accessors: guarantee global_load/global_store (LOADcnt/STOREcnt
// only) instead of flat_load (which also ticks DScnt and serializes against
// our LDS B-fragment traffic).
__device__ __forceinline__ float gloadf(const float* p) {
  return *(AS1 const float*)(uptr)p;
}
__device__ __forceinline__ void gstoref(float* p, float v) {
  *(AS1 float*)(uptr)p = v;
}
__device__ __forceinline__ v16bf gload_frag(const unsigned short* p) {
  return *(AS1 const v16bf*)(uptr)p;
}

__device__ __forceinline__ unsigned int f2bf1(float f) {
  unsigned int u = __float_as_uint(f);
  return (u + 0x7FFFu + ((u >> 16) & 1u)) >> 16;   // RNE to bf16
}
__device__ __forceinline__ unsigned int pack2bf(float a, float b) {
  return f2bf1(a) | (f2bf1(b) << 16);
}

// ---------------------------------------------------------------------------
// Pack fp32 weights [192,384] -> bf16 WMMA A-fragment-linear layout:
// wp[gate][mt][kt][lane][i], where lane l holds A[m,k]:
//   m = mt*16 + (l&15),  k = kt*32 + 16*(l>>4) + i
// (contiguous k order; LDS B fragments use the same)
// ---------------------------------------------------------------------------
__global__ void pack_weights_kernel(const float* __restrict__ wz,
                                    const float* __restrict__ wr,
                                    const float* __restrict__ wh,
                                    unsigned short* __restrict__ wp) {
  int idx = blockIdx.x * blockDim.x + threadIdx.x;
  if (idx >= 3 * GATE_STRIDE) return;
  int gate = idx / GATE_STRIDE;
  int rem  = idx % GATE_STRIDE;
  int mt   = rem / (KT_ * 32 * 16);
  int r2   = rem % (KT_ * 32 * 16);
  int kt   = r2 / (32 * 16);
  int r3   = r2 % (32 * 16);
  int lane = r3 >> 4;
  int i    = r3 & 15;
  int m = mt * 16 + (lane & 15);
  int k = kt * 32 + 16 * (lane >> 4) + i;
  const float* w = (gate == 0) ? wz : ((gate == 1) ? wr : wh);
  wp[idx] = (unsigned short)f2bf1(w[m * C2 + k]);
}

// ---------------------------------------------------------------------------
// Main ConvGRU kernel. One workgroup = 32 pixels, 6 waves.
// Wave w owns M-tiles {w, w+6} (channels [16w,16w+16) u [16(w+6),...)) for
// all 3 gates -> z and h state stay in this wave's registers for all t.
// B-fragment LDS layout (contiguous k, same order as packed A): bfrag[ncol][kt][lane][i] bf16
// where lane = n + 16*(k>=16), i = k&15 within the 32-wide K tile.
// ---------------------------------------------------------------------------
__global__ __launch_bounds__(BLOCK) void convgru_kernel(
    const float* __restrict__ video,
    const float* __restrict__ bzg,
    const float* __restrict__ brg,
    const float* __restrict__ bhg,
    const unsigned short* __restrict__ wp,
    float* __restrict__ out)
{
  __shared__ __align__(32) unsigned short bfrag[2 * KT_ * 32 * 16]; // 24 KB
  __shared__ float lbias[3 * CC];                                   // 2.25 KB

  const int tid  = threadIdx.x;
  const int w    = tid >> 5;      // wave id 0..5
  const int lane = tid & 31;
  const int hi   = lane >> 4;
  const int nn   = lane & 15;

  const int pbase = blockIdx.x * TILEN;
  const int b  = pbase / HW_;     // 3136 % 32 == 0: tile never crosses batch
  const int sp = pbase % HW_;

  // biases -> LDS (BLOCK == CC, one element per gate per thread)
  lbias[tid]          = bzg[tid];
  lbias[CC + tid]     = brg[tid];
  lbias[2 * CC + tid] = bhg[tid];

  // zero h part of B buffer (kt 6..11, both ncols): h_{-1} = 0
  {
    unsigned int* b32 = (unsigned int*)bfrag;
#pragma unroll
    for (int nc = 0; nc < 2; ++nc) {
      const int base = ((nc * KT_ + 6) * 32 * 16) >> 1;
      for (int j = tid; j < (6 * 32 * 16) / 2; j += BLOCK) b32[base + j] = 0u;
    }
  }

  v8f hreg[2][2];                  // fp32 state: [tile][ncol]
#pragma unroll
  for (int tt = 0; tt < 2; ++tt)
#pragma unroll
    for (int n = 0; n < 2; ++n) hreg[tt][n] = (v8f)0.0f;

  for (int t = 0; t < T_; ++t) {
    // Opaque zero offset, redefined every timestep: defeats LICM on the 72
    // per-step A-fragment loads (a hoist previously spilled 288 VGPRs to
    // scratch) while keeping the base pointer's global address space intact.
    unsigned int zoff = 0u;
    asm volatile("" : "+s"(zoff));
    const unsigned short* wpz = wp + zoff;
    const unsigned short* wpr = wp + GATE_STRIDE + zoff;
    const unsigned short* wph = wp + 2 * GATE_STRIDE + zoff;

    // ---- stage x_t (fp32 global -> bf16 LDS B-fragments, kt = w) ----
    const float* xsrc = video + (size_t)((b * T_ + t) * CC) * HW_ + sp;
    const int c0 = w * 32 + 16 * hi;   // 16 consecutive channels per thread
#pragma unroll
    for (int nc = 0; nc < 2; ++nc) {
      unsigned int pk[8];
#pragma unroll
      for (int i = 0; i < 16; i += 2) {
        float f0 = gloadf(&xsrc[(size_t)(c0 + i    ) * HW_ + nc * 16 + nn]);
        float f1 = gloadf(&xsrc[(size_t)(c0 + i + 1) * HW_ + nc * 16 + nn]);
        pk[i >> 1] = pack2bf(f0, f1);
      }
      uint4* dst = (uint4*)&bfrag[((nc * KT_ + w) * 32 + lane) * 16];
      dst[0] = make_uint4(pk[0], pk[1], pk[2], pk[3]);
      dst[1] = make_uint4(pk[4], pk[5], pk[6], pk[7]);
    }
    __syncthreads();   // x_t + h_{t-1} fragments visible

    // ---- phase 1: Z and R GEMMs over K=384 ----
    v8f accZ[2][2], accR[2][2];
#pragma unroll
    for (int tt = 0; tt < 2; ++tt)
#pragma unroll
      for (int n = 0; n < 2; ++n) { accZ[tt][n] = (v8f)0.0f; accR[tt][n] = (v8f)0.0f; }

    for (int kt = 0; kt < KT_; ++kt) {
      v16bf bf0 = *(const v16bf*)&bfrag[((0 * KT_ + kt) * 32 + lane) * 16];
      v16bf bf1 = *(const v16bf*)&bfrag[((1 * KT_ + kt) * 32 + lane) * 16];
#pragma unroll
      for (int tt = 0; tt < 2; ++tt) {
        const int mt = w + 6 * tt;
        v16bf az = gload_frag(&wpz[((mt * KT_ + kt) * 32 + lane) * 16]);
        v16bf ar = gload_frag(&wpr[((mt * KT_ + kt) * 32 + lane) * 16]);
        accZ[tt][0] = __builtin_amdgcn_wmma_f32_16x16x32_bf16(false, az, false, bf0, (short)0, accZ[tt][0], false, false);
        accZ[tt][1] = __builtin_amdgcn_wmma_f32_16x16x32_bf16(false, az, false, bf1, (short)0, accZ[tt][1], false, false);
        accR[tt][0] = __builtin_amdgcn_wmma_f32_16x16x32_bf16(false, ar, false, bf0, (short)0, accR[tt][0], false, false);
        accR[tt][1] = __builtin_amdgcn_wmma_f32_16x16x32_bf16(false, ar, false, bf1, (short)0, accR[tt][1], false, false);
      }
    }
    __syncthreads();   // all waves done reading h-part before r*h overwrite

    // ---- sigmoid(z), sigmoid(r); scatter r*h (bf16) into kt 6..11 ----
#pragma unroll
    for (int tt = 0; tt < 2; ++tt) {
      const int mt  = w + 6 * tt;
      const int kth = 6 + (mt >> 1);
#pragma unroll
      for (int n = 0; n < 2; ++n) {
        v8f zv; float rh[8];
#pragma unroll
        for (int i = 0; i < 8; ++i) {
          const int ch = mt * 16 + i + 8 * hi;   // C/D layout: M = reg + 8*hi
          float zz = accZ[tt][n][i] + lbias[ch];
          float rr = accR[tt][n][i] + lbias[CC + ch];
          zz = 1.0f / (1.0f + __expf(-zz));
          rr = 1.0f / (1.0f + __expf(-rr));
          zv[i] = zz;
          rh[i] = rr * hreg[tt][n][i];
        }
        accZ[tt][n] = zv;                         // keep z for final update
        unsigned int pk[4];
#pragma unroll
        for (int i = 0; i < 8; i += 2) pk[i >> 1] = pack2bf(rh[i], rh[i + 1]);
        // target: XH channel gc = mt*16 + i + 8*hi -> lane nn+16*(mt&1), i = r+8*hi
        uint4* dst = (uint4*)&bfrag[(((n * KT_ + kth) * 32) + nn + 16 * (mt & 1)) * 16 + 8 * hi];
        *dst = make_uint4(pk[0], pk[1], pk[2], pk[3]);
      }
    }
    __syncthreads();   // r*h fragments visible

    // ---- phase 2: H candidate GEMM over [x ; r*h] ----
    v8f accH[2][2];
#pragma unroll
    for (int tt = 0; tt < 2; ++tt)
#pragma unroll
      for (int n = 0; n < 2; ++n) accH[tt][n] = (v8f)0.0f;

    for (int kt = 0; kt < KT_; ++kt) {
      v16bf bf0 = *(const v16bf*)&bfrag[((0 * KT_ + kt) * 32 + lane) * 16];
      v16bf bf1 = *(const v16bf*)&bfrag[((1 * KT_ + kt) * 32 + lane) * 16];
#pragma unroll
      for (int tt = 0; tt < 2; ++tt) {
        const int mt = w + 6 * tt;
        v16bf ah = gload_frag(&wph[((mt * KT_ + kt) * 32 + lane) * 16]);
        accH[tt][0] = __builtin_amdgcn_wmma_f32_16x16x32_bf16(false, ah, false, bf0, (short)0, accH[tt][0], false, false);
        accH[tt][1] = __builtin_amdgcn_wmma_f32_16x16x32_bf16(false, ah, false, bf1, (short)0, accH[tt][1], false, false);
      }
    }
    __syncthreads();   // done reading before h_new overwrite of kt 6..11

    // ---- GRU update: h = (1-z)h + z*tanh(acc+bh); write out + stage h bf16 ----
    float* odst = out + (size_t)((b * T_ + t) * CC) * HW_ + sp;
#pragma unroll
    for (int tt = 0; tt < 2; ++tt) {
      const int mt  = w + 6 * tt;
      const int kth = 6 + (mt >> 1);
#pragma unroll
      for (int n = 0; n < 2; ++n) {
        v8f hn;
#pragma unroll
        for (int i = 0; i < 8; ++i) {
          const int ch = mt * 16 + i + 8 * hi;
          float e  = __expf(2.0f * (accH[tt][n][i] + lbias[2 * CC + ch]));
          float th = 1.0f - 2.0f / (e + 1.0f);    // tanh, inf-safe
          float z  = accZ[tt][n][i];
          float h0 = hreg[tt][n][i];
          hn[i] = h0 + z * (th - h0);
        }
        hreg[tt][n] = hn;
#pragma unroll
        for (int i = 0; i < 8; ++i) {
          const int ch = mt * 16 + i + 8 * hi;
          gstoref(&odst[(size_t)ch * HW_ + n * 16 + nn], hn[i]);
        }
        unsigned int pk[4];
#pragma unroll
        for (int i = 0; i < 8; i += 2) pk[i >> 1] = pack2bf(hn[i], hn[i + 1]);
        uint4* dst = (uint4*)&bfrag[(((n * KT_ + kth) * 32) + nn + 16 * (mt & 1)) * 16 + 8 * hi];
        *dst = make_uint4(pk[0], pk[1], pk[2], pk[3]);
      }
    }
    __syncthreads();   // h_t fragments ready; safe to overwrite x-part next t
  }
}

// ---------------------------------------------------------------------------
extern "C" void kernel_launch(void* const* d_in, const int* in_sizes, int n_in,
                              void* d_out, int out_size, void* d_ws, size_t ws_size,
                              hipStream_t stream) {
  const float* video = (const float*)d_in[0];
  const float* w_z   = (const float*)d_in[1];
  const float* b_z   = (const float*)d_in[2];
  const float* w_r   = (const float*)d_in[3];
  const float* b_r   = (const float*)d_in[4];
  const float* w_h   = (const float*)d_in[5];
  const float* b_h   = (const float*)d_in[6];

  unsigned short* wpack = (unsigned short*)d_ws;   // 3*73728 bf16 = 432 KB

  const int pack_elems = 3 * GATE_STRIDE;
  pack_weights_kernel<<<(pack_elems + 255) / 256, 256, 0, stream>>>(w_z, w_r, w_h, wpack);

  convgru_kernel<<<NPIX / TILEN, BLOCK, 0, stream>>>(
      video, b_z, b_r, b_h, wpack, (float*)d_out);
}
